// subnet_flex_recurrent_48756468744412
// MI455X (gfx1250) — compile-verified, matrix operand order checked by reading, not run on hardware
//
#include <hip/hip_runtime.h>
#include <hip/hip_bf16.h>

typedef _Float16 v16h __attribute__((ext_vector_type(16)));
typedef float    v8f  __attribute__((ext_vector_type(8)));

#define NB     50
#define S51    51
#define B256   256
#define HID    128
#define NROWS  (B256 * S51 * S51)        // 665856
#define ROWS_PER_WG 128
#define NWG    (NROWS / ROWS_PER_WG)     // 5202
#define LDST   136                       // padded LDS row stride in halfs

// ---------------------------------------------------------------- tables
__global__ void k_tables(const float* __restrict__ W1,
                         float* __restrict__ chalf, float* __restrict__ ccw,
                         float* __restrict__ w1c0) {
  int j = threadIdx.x;
  if (j < HID) w1c0[j] = W1[j * 64];     // W1[:,0]
  if (j <= NB) {
    const double pi = 3.14159265358979323846;
    chalf[j] = (float)((cos((double)j * pi / (double)NB) + 1.0) * 0.5);
    double acc = 0.0;
    for (int i = 0; i <= NB; ++i) {
      double Wi = (i == 0) ? 1.0 : ((i & 1) ? 0.0 : 2.0 / (1.0 - (double)(i * i)));
      double L;
      if (j == 0)       L = 0.5;
      else if (j == NB) L = 0.5 * cos((double)i * pi);
      else              L = cos((double)(i * j) * pi / (double)NB);
      acc += L * Wi;
    }
    ccw[j] = (float)(acc * (2.0 / (double)NB));
  }
}

// ---------------------------------------------------------------- W2 -> f16
__global__ void k_w2h(const float* __restrict__ W2, _Float16* __restrict__ W2h) {
  int i = blockIdx.x * blockDim.x + threadIdx.x;
  if (i < HID * HID) W2h[i] = (_Float16)W2[i];
}

// ------------------------------------------- xm[b] and hW1[b][k] precompute
__global__ __launch_bounds__(128) void k_xm_hw1(
    const float* __restrict__ x_, const float* __restrict__ h_,
    const float* __restrict__ W1, const float* __restrict__ b1,
    const float* __restrict__ lw, float* __restrict__ xm,
    float* __restrict__ hW1) {
  int b = blockIdx.x, k = threadIdx.x;
  float acc = b1[k];
  const float* hb = h_ + b * 63;
  const float* w  = W1 + k * 64 + 1;
  #pragma unroll 7
  for (int i = 0; i < 63; ++i) acc = fmaf(hb[i], w[i], acc);
  hW1[b * HID + k] = acc;

  __shared__ float red[128];
  float p = 0.f;
  for (int j = k; j < 512; j += 128) p = fmaf(x_[b * 512 + j], lw[j], p);
  red[k] = p; __syncthreads();
  for (int off = 64; off > 0; off >>= 1) {
    if (k < off) red[k] += red[k + off];
    __syncthreads();
  }
  if (k == 0) xm[b] = red[0];
}

// -------------------------------------- per-step: xT, xmax, wfac, zero accum
__global__ __launch_bounds__(256) void k_setup(
    const float* __restrict__ xm, const float* __restrict__ sig,
    const float* __restrict__ chalf, const float* __restrict__ ccw,
    float* __restrict__ xT, float* __restrict__ Xa, float* __restrict__ wfac,
    float* __restrict__ xmaxp, float* __restrict__ out_acc, int first) {
  __shared__ float red[256];
  int b = threadIdx.x;
  float lam = 1.f / (1.f + __expf(-sig[0]));
  float prev = first ? 0.f : out_acc[b];
  float xt = (1.f - lam) * xm[b] + lam * prev;
  xT[b] = xt;
  red[b] = xt; __syncthreads();
  for (int off = 128; off > 0; off >>= 1) {
    if (b < off) red[b] = fmaxf(red[b], red[b + off]);
    __syncthreads();
  }
  // c in [0,1] with both endpoints present -> global max X = max(0, max xT)
  float xmax = fmaxf(red[0], 0.f) + 10.f;
  if (b == 0) *xmaxp = xmax;
  out_acc[b] = 0.f;
  for (int so = 0; so < S51; ++so) {
    float X = xt * chalf[so];
    Xa[b * S51 + so]   = X;
    wfac[b * S51 + so] = (xmax - X) * 0.25f * ccw[so] * xt;
  }
}

// ------------------------------------------------------- the big WMMA kernel
__global__ __launch_bounds__(256) void k_big(
    const float* __restrict__ Xa, const float* __restrict__ wfac,
    const float* __restrict__ hW1, const _Float16* __restrict__ W2h,
    const float* __restrict__ chalf, const float* __restrict__ ccw,
    const float* __restrict__ xmaxp, const float* __restrict__ w1c0,
    const float* __restrict__ b2, const float* __restrict__ W3,
    const float* __restrict__ b3, const float* __restrict__ final_w,
    float* __restrict__ out_acc) {
  __shared__ _Float16 lds_a1[ROWS_PER_WG * LDST];   // ~34 KB
  __shared__ _Float16 lds_w2[HID * LDST];           // ~34 KB (W2 staged once)

  const int tid = threadIdx.x;
  const int wg  = blockIdx.x;
  const float xmax = *xmaxp;

  // ---- Phase A: build a1 tile (128 rows x 128 neurons) in f16, in LDS
  {
    int r  = tid >> 1;
    int c0 = (tid & 1) * 64;
    int R  = wg * ROWS_PER_WG + r;
    int b  = R / 2601;  int rem = R - b * 2601;
    int so = rem / S51; int si  = rem - so * S51;
    float X = Xa[b * S51 + so];
    float t = X + (xmax - X) * chalf[si];
    const float4* hv = (const float4*)(hW1 + b * HID + c0);
    const float4* wv = (const float4*)(w1c0 + c0);
    #pragma unroll
    for (int k4 = 0; k4 < 16; ++k4) {
      float4 hw = hv[k4];
      float4 wc = wv[k4];
      float a0 = fmaxf(fmaf(t, wc.x, hw.x), 0.f);
      float a1 = fmaxf(fmaf(t, wc.y, hw.y), 0.f);
      float a2 = fmaxf(fmaf(t, wc.z, hw.z), 0.f);
      float a3 = fmaxf(fmaf(t, wc.w, hw.w), 0.f);
      union { _Float16 h[4]; uint2 u; } pk;
      pk.h[0] = (_Float16)a0; pk.h[1] = (_Float16)a1;
      pk.h[2] = (_Float16)a2; pk.h[3] = (_Float16)a3;
      *(uint2*)&lds_a1[r * LDST + c0 + k4 * 4] = pk.u;
    }
  }

  // ---- Phase A2: cooperative stage of W2 (f16, row-major) into LDS
  {
    int row = tid >> 1;
    int h64 = (tid & 1) * 64;
    const uint4* src = (const uint4*)(W2h + row * HID + h64);  // 8 halfs/uint4
    uint4* dst = (uint4*)&lds_w2[row * LDST + h64];
    #pragma unroll
    for (int j = 0; j < 8; ++j) dst[j] = src[j];
  }
  __syncthreads();

  // ---- Phase B: 8 waves, each owns a 16-row strip; GEMM a1 @ W2^T via WMMA
  const int wv   = tid >> 5;       // wave id 0..7
  const int lane = tid & 31;
  const int lr   = lane & 15;      // row-in-tile (A) / col-in-tile (B,C)
  const int hg   = lane >> 4;      // half-wave group

  // A fragments: 16x32 f16 tile layout (lanes 0-15: K=0..7,16..23; 16-31: +8)
  v16h afr[4];
  #pragma unroll
  for (int kt = 0; kt < 4; ++kt) {
    const _Float16* base = &lds_a1[(wv * 16 + lr) * LDST + kt * 32 + hg * 8];
    union { uint4 u[2]; v16h v; } f;
    f.u[0] = *(const uint4*)base;
    f.u[1] = *(const uint4*)(base + 16);
    afr[kt] = f.v;
  }

  float p3[8];
  #pragma unroll
  for (int v = 0; v < 8; ++v) p3[v] = 0.f;

  #pragma unroll
  for (int nt = 0; nt < 8; ++nt) {
    v8f acc = {0.f, 0.f, 0.f, 0.f, 0.f, 0.f, 0.f, 0.f};
    const int n = nt * 16 + lr;
    #pragma unroll
    for (int kt = 0; kt < 4; ++kt) {
      // B fragment: column n of B == row n of W2 (out = a1 @ W2^T),
      // same K packing as A; served from LDS
      const _Float16* base = &lds_w2[n * LDST + kt * 32 + hg * 8];
      union { uint4 u[2]; v16h v; } f;
      f.u[0] = *(const uint4*)base;
      f.u[1] = *(const uint4*)(base + 16);
      acc = __builtin_amdgcn_wmma_f32_16x16x32_f16(
          false, afr[kt], false, f.v, (short)0, acc, false, false);
    }
    float b2n = b2[n];
    float w3n = W3[n];
    #pragma unroll
    for (int v = 0; v < 8; ++v) {
      float a2 = fmaxf(acc[v] + b2n, 0.f);   // relu(layer2)
      p3[v] = fmaf(a2, w3n, p3[v]);          // layer-3 partial dot
    }
  }

  // reduce the layer-3 dot across the 16 lanes holding the same row
  #pragma unroll
  for (int v = 0; v < 8; ++v) {
    float s = p3[v];
    s += __shfl_xor(s, 1, 32);
    s += __shfl_xor(s, 2, 32);
    s += __shfl_xor(s, 4, 32);
    s += __shfl_xor(s, 8, 32);
    p3[v] = s;
  }

  // tail: lanes 0 (rows 0..7) and 16 (rows 8..15) of each wave
  if (lr == 0) {
    const float fb3 = b3[0];
    const float fw0 = final_w[0], fw1 = final_w[1], fw2 = final_w[2],
                fw3 = final_w[3], fw4 = final_w[4];
    float accum = 0.f; int cur_b = -1;
    #pragma unroll
    for (int v = 0; v < 8; ++v) {
      int m   = wv * 16 + hg * 8 + v;            // C layout: M = v + 8*(lane/16)
      int R   = wg * ROWS_PER_WG + m;
      int b   = R / 2601;  int rem = R - b * 2601;
      int so  = rem / S51; int si  = rem - so * S51;
      float d  = p3[v] + fb3;
      float zz = (d > 0.f) ? (d + 1.f) : __expf(d);   // elu(d)+1, zz > 0
      float sq = sqrtf(zz);
      float z2 = zz * zz;
      float q1 = zz * sq, q3 = z2 * sq, q4 = z2 * zz, q5 = q4 * sq;
      float g = fw0 / (q1 + 1.f) + fw1 / (z2 + 1.f) + fw2 / (q3 + 1.f) +
                fw3 / (q4 + 1.f) + fw4 / (q5 + 1.f);
      g = fmaxf(g, 0.f);
      float contrib = g * ccw[si] * wfac[b * S51 + so];
      if (b != cur_b) {
        if (cur_b >= 0) atomicAdd(&out_acc[cur_b], accum);
        cur_b = b; accum = contrib;
      } else {
        accum += contrib;
      }
    }
    if (cur_b >= 0) atomicAdd(&out_acc[cur_b], accum);
  }
}

// ---------------------------------------------------------------- finalize
__global__ void k_copy(const float* __restrict__ src, float* __restrict__ dst) {
  dst[threadIdx.x] = src[threadIdx.x];
}

extern "C" void kernel_launch(void* const* d_in, const int* in_sizes, int n_in,
                              void* d_out, int out_size, void* d_ws, size_t ws_size,
                              hipStream_t stream) {
  const float* x_   = (const float*)d_in[0];
  const float* h_   = (const float*)d_in[1];
  const float* W1   = (const float*)d_in[2];
  const float* b1   = (const float*)d_in[3];
  const float* W2   = (const float*)d_in[4];
  const float* b2   = (const float*)d_in[5];
  const float* W3   = (const float*)d_in[6];
  const float* b3   = (const float*)d_in[7];
  const float* fwv  = (const float*)d_in[8];
  const float* lw   = (const float*)d_in[9];
  const float* sig  = (const float*)d_in[10];
  float* out = (float*)d_out;

  // workspace layout (floats), base is 256B aligned from hipMalloc
  float* ws     = (float*)d_ws;
  float* xm     = ws;                        // 256
  float* hW1    = ws + 256;                  // 32768
  _Float16* W2h = (_Float16*)(ws + 256 + 32768);  // 16384 halfs (8192 floats)
  float* xT     = ws + 256 + 32768 + 8192;   // 256
  float* Xa     = xT + 256;                  // 13056
  float* wfac   = Xa + 13056;                // 13056
  float* outac  = wfac + 13056;              // 256
  float* xmaxp  = outac + 256;               // 1 (+pad)
  float* ccw    = xmaxp + 16;                // 51 (+pad)
  float* chalf  = ccw + 64;                  // 51 (+pad)
  float* w1c0   = chalf + 64;                // 128
  (void)in_sizes; (void)n_in; (void)out_size; (void)ws_size;

  hipLaunchKernelGGL(k_tables, dim3(1), dim3(128), 0, stream, W1, chalf, ccw, w1c0);
  hipLaunchKernelGGL(k_w2h, dim3(64), dim3(256), 0, stream, W2, W2h);
  hipLaunchKernelGGL(k_xm_hw1, dim3(B256), dim3(128), 0, stream,
                     x_, h_, W1, b1, lw, xm, hW1);

  for (int step = 0; step < 2; ++step) {
    hipLaunchKernelGGL(k_setup, dim3(1), dim3(256), 0, stream,
                       xm, sig, chalf, ccw, xT, Xa, wfac, xmaxp, outac,
                       (step == 0) ? 1 : 0);
    hipLaunchKernelGGL(k_big, dim3(NWG), dim3(256), 0, stream,
                       Xa, wfac, hW1, W2h, chalf, ccw, xmaxp, w1c0,
                       b2, W3, b3, fwv, outac);
  }
  hipLaunchKernelGGL(k_copy, dim3(1), dim3(256), 0, stream, outac, out);
}
